// G2IrrepTower_43224550867149
// MI455X (gfx1250) — matrix operand order repaired and checked by reading, not compiled, not run on hardware
//
#include <hip/hip_runtime.h>
#include <hip/hip_bf16.h>
#include <math.h>

typedef __attribute__((ext_vector_type(16))) __bf16 v16bf;
typedef __attribute__((ext_vector_type(8)))  float  v8f;

#define BROWS 65536
#define KPMAX 480

// ---------------- bf16 helpers ----------------
__device__ __forceinline__ __bf16 f2bf(float f) {
    unsigned u = __builtin_bit_cast(unsigned, f);
    unsigned r = u + 0x7FFFu + ((u >> 16) & 1u);
    unsigned short h = (unsigned short)(r >> 16);
    return __builtin_bit_cast(__bf16, h);
}
__device__ __forceinline__ float bf2f(__bf16 b) {
    unsigned short h = __builtin_bit_cast(unsigned short, b);
    unsigned u = ((unsigned)h) << 16;
    return __builtin_bit_cast(float, u);
}
__device__ __forceinline__ float geluf(float x) {
    return 0.5f * x * (1.0f + erff(x * 0.70710678118654752f));
}
__device__ __forceinline__ void ln_inplace(float* v, int n, const float* g, const float* b) {
    float m = 0.f;
    for (int i = 0; i < n; ++i) m += v[i];
    m /= (float)n;
    float var = 0.f;
    for (int i = 0; i < n; ++i) { float d = v[i] - m; var += d * d; }
    var /= (float)n;
    float is = rsqrtf(var + 1e-5f);
    for (int i = 0; i < n; ++i) v[i] = (v[i] - m) * is * g[i] + b[i];
}

// ---------------- workspace layout (bytes) ----------------
// const block (floats): PHI 343 | B14 686 | S27 1323 | P64 6272 | P77A 15092 | P77B 15092 | KER 294
#define OFF_PHI  0
#define OFF_B14  343
#define OFF_S27  1029
#define OFF_P64  2352
#define OFF_P77A 8624
#define OFF_P77B 23716
#define OFF_KER  38808

static constexpr size_t O_CST  = 0;
static constexpr size_t O_WHPB = 163840;                                   // [192,288] bf16
static constexpr size_t O_WQB  = O_WHPB + (size_t)192 * 288 * 2;           // [64,64] bf16
static constexpr size_t O_WKB  = O_WQB + 64 * 64 * 2;
static constexpr size_t O_WVB  = O_WKB + 64 * 64 * 2;
static constexpr size_t O_WOB  = O_WVB + 64 * 64 * 2;
static constexpr size_t O_W1B  = O_WOB + 64 * 64 * 2;                      // [256,480] bf16
static constexpr size_t O_W2B  = O_W1B + (size_t)256 * 480 * 2;            // [128,256] bf16
static constexpr size_t O_MIX  = O_W2B + (size_t)128 * 256 * 2;            // [B,8] f32
static constexpr size_t O_LOWF = O_MIX + (size_t)BROWS * 8 * 4;            // [B,272] f32
static constexpr size_t O_LOWB = O_LOWF + (size_t)BROWS * 272 * 4;         // [B,288] bf16
static constexpr size_t O_H189 = O_LOWB + (size_t)BROWS * 288 * 2;         // [B,192] f32
static constexpr size_t O_PAD  = O_H189 + (size_t)BROWS * 192 * 4;         // [B,512] bf16
static constexpr size_t O_Q    = O_PAD + (size_t)BROWS * 512 * 2;          // [8B,64] bf16
static constexpr size_t O_KB   = O_Q + (size_t)BROWS * 512 * 2;
static constexpr size_t O_V    = O_KB + (size_t)BROWS * 512 * 2;
static constexpr size_t O_AO   = O_V + (size_t)BROWS * 512 * 2;
static constexpr size_t O_ATT  = O_AO + (size_t)BROWS * 512 * 2;           // [B,512] f32
static constexpr size_t O_X1   = O_ATT + (size_t)BROWS * 512 * 4;          // [B,480] bf16
static constexpr size_t O_PRE1 = O_X1 + (size_t)BROWS * 480 * 2;           // [B,256] f32
static constexpr size_t O_H256 = O_PRE1 + (size_t)BROWS * 256 * 4;         // [B,256] bf16

// ---------------- constant construction ----------------
__device__ unsigned long long lcg_next(unsigned long long& s) {
    s = s * 6364136223846793005ULL + 1442695040888963407ULL;
    return s;
}
__device__ float frand(unsigned long long& s) {
    return ((float)((lcg_next(s) >> 40) & 0xFFFFFF) / 8388608.0f) - 1.0f;
}
__device__ void gen_orth(float* dst, int R, int K, unsigned long long seed) {
    unsigned long long s = seed;
    for (int r = 0; r < R; ++r) {
        float* v = dst + (size_t)r * K;
        for (int k = 0; k < K; ++k) v[k] = frand(s);
#pragma clang loop unroll(disable)
        for (int p = 0; p < r; ++p) {
            const float* u = dst + (size_t)p * K;
            float d = 0.f;
            for (int k = 0; k < K; ++k) d += v[k] * u[k];
            for (int k = 0; k < K; ++k) v[k] -= d * u[k];
        }
        float nn = 0.f;
        for (int k = 0; k < K; ++k) nn += v[k] * v[k];
        float sc = rsqrtf(nn + 1e-12f);
        for (int k = 0; k < K; ++k) v[k] *= sc;
    }
}

__global__ void k_setup(float* cst) {
    if (threadIdx.x != 0 || blockIdx.x != 0) return;
    float* PHI = cst + OFF_PHI;
    for (int t = 0; t < 343; ++t) PHI[t] = 0.f;
    const int tri[7][3] = {{0,1,2},{0,3,4},{0,5,6},{1,3,5},{1,6,4},{2,3,6},{2,4,5}};
    for (int t = 0; t < 7; ++t) {
        int i = tri[t][0], j = tri[t][1], k = tri[t][2];
        PHI[i*49+j*7+k] = 1.f; PHI[j*49+k*7+i] = 1.f; PHI[k*49+i*7+j] = 1.f;
        PHI[j*49+i*7+k] = -1.f; PHI[i*49+k*7+j] = -1.f; PHI[k*49+j*7+i] = -1.f;
    }
    int pi[21], pj[21];
    {
        int m = 0;
        for (int i = 0; i < 7; ++i)
            for (int j = i + 1; j < 7; ++j) { pi[m] = i; pj[m] = j; ++m; }
    }
    float M[7][21];
    for (int r = 0; r < 7; ++r)
        for (int p = 0; p < 21; ++p)
            M[r][p] = 1.41421356237f * PHI[pi[p]*49 + pj[p]*7 + r];
    float Aug[7][14];
    for (int i = 0; i < 7; ++i)
        for (int j = 0; j < 7; ++j) {
            float s = 0.f;
            for (int p = 0; p < 21; ++p) s += M[i][p] * M[j][p];
            Aug[i][j] = s;
            Aug[i][7 + j] = (i == j) ? 1.f : 0.f;
        }
#pragma clang loop unroll(disable)
    for (int col = 0; col < 7; ++col) {
        int piv = col; float best = fabsf(Aug[col][col]);
        for (int r = col + 1; r < 7; ++r)
            if (fabsf(Aug[r][col]) > best) { best = fabsf(Aug[r][col]); piv = r; }
        if (piv != col)
            for (int j = 0; j < 14; ++j) { float t = Aug[col][j]; Aug[col][j] = Aug[piv][j]; Aug[piv][j] = t; }
        float ip = 1.f / Aug[col][col];
        for (int j = 0; j < 14; ++j) Aug[col][j] *= ip;
        for (int r = 0; r < 7; ++r) {
            if (r == col) continue;
            float f = Aug[r][col];
            for (int j = 0; j < 14; ++j) Aug[r][j] -= f * Aug[col][j];
        }
    }
    float* KER = cst + OFF_KER;
    int cnt = 0;
#pragma clang loop unroll(disable)
    for (int p = 0; p < 21 && cnt < 14; ++p) {
        float u[7];
        for (int r = 0; r < 7; ++r) {
            float s = 0.f;
            for (int c = 0; c < 7; ++c) s += Aug[r][7 + c] * M[c][p];
            u[r] = s;
        }
        float v[21];
        for (int q = 0; q < 21; ++q) {
            float s = (q == p) ? 1.f : 0.f;
            for (int r = 0; r < 7; ++r) s -= M[r][q] * u[r];
            v[q] = s;
        }
        for (int pr = 0; pr < cnt; ++pr) {
            const float* w = KER + pr * 21;
            float d = 0.f;
            for (int q = 0; q < 21; ++q) d += v[q] * w[q];
            for (int q = 0; q < 21; ++q) v[q] -= d * w[q];
        }
        float nn = 0.f;
        for (int q = 0; q < 21; ++q) nn += v[q] * v[q];
        if (nn > 1e-6f) {
            float sc = rsqrtf(nn);
            for (int q = 0; q < 21; ++q) KER[cnt * 21 + q] = v[q] * sc;
            ++cnt;
        }
    }
    float* B14 = cst + OFF_B14;
    for (int t = 0; t < 686; ++t) B14[t] = 0.f;
    for (int m = 0; m < 14; ++m)
        for (int p = 0; p < 21; ++p) {
            float c = KER[m * 21 + p] * 0.70710678118f;
            B14[m*49 + pi[p]*7 + pj[p]] += c;
            B14[m*49 + pj[p]*7 + pi[p]] -= c;
        }
    float* S27 = cst + OFF_S27;
    for (int t = 0; t < 1323; ++t) S27[t] = 0.f;
    for (int p = 0; p < 21; ++p) {
        S27[p*49 + pi[p]*7 + pj[p]] = 0.70710678118f;
        S27[p*49 + pj[p]*7 + pi[p]] = 0.70710678118f;
    }
    float Dd[6][7];
    cnt = 0;
    for (int i = 0; i < 7 && cnt < 6; ++i) {
        float v[7];
        for (int j = 0; j < 7; ++j) v[j] = ((j == i) ? 1.f : 0.f) - 1.f / 7.f;
        for (int p = 0; p < cnt; ++p) {
            float d = 0.f;
            for (int j = 0; j < 7; ++j) d += v[j] * Dd[p][j];
            for (int j = 0; j < 7; ++j) v[j] -= d * Dd[p][j];
        }
        float nn = 0.f;
        for (int j = 0; j < 7; ++j) nn += v[j] * v[j];
        if (nn > 1e-8f) {
            float sc = rsqrtf(nn);
            for (int j = 0; j < 7; ++j) Dd[cnt][j] = v[j] * sc;
            ++cnt;
        }
    }
    for (int m = 0; m < 6; ++m)
        for (int j = 0; j < 7; ++j)
            S27[(21 + m)*49 + j*7 + j] = Dd[m][j];
    gen_orth(cst + OFF_P64, 64, 98, 0x12345678ULL);
    gen_orth(cst + OFF_P77A, 77, 196, 0x9E3779B9ULL);
    gen_orth(cst + OFF_P77B, 77, 196, 0xDEADBEEFULL);
}

// ---------------- weight cast/pad ----------------
__global__ void k_castpad(const float* __restrict__ src, __bf16* __restrict__ dst,
                          int N, int K, int Np, int Kp) {
    int t = blockIdx.x * blockDim.x + threadIdx.x;
    if (t >= Np * Kp) return;
    int n = t / Kp, k = t % Kp;
    float v = (n < N && k < K) ? src[n * K + k] : 0.f;
    dst[t] = f2bf(v);
}

// ---------------- WMMA GEMM: C[M,N] = A[M,K] * W[N,K]^T (+bias) ----------------
// Block = 8 waves; each wave owns a 16x32 output tile (2 accumulators).
// Weight tile (32 cols x K) staged into LDS via CDNA5 async global->LDS loads.
__global__ void __launch_bounds__(256)
k_gemm_bf16(const __bf16* __restrict__ A, int lda,
            const __bf16* __restrict__ W, int ldw,
            const float* __restrict__ bias,
            void* __restrict__ Cout, int ldc,
            int K, int Nvalid, int outBf16) {
    __shared__ __bf16 shW[32 * KPMAX];
    int tid = threadIdx.x;
    int lane = tid & 31;
    int wave = tid >> 5;
    int row0 = (blockIdx.y * 8 + wave) * 16;
    int col0 = blockIdx.x * 32;
    int mrow = lane & 15;
    int kh = lane >> 4;

    // --- stage W[col0..col0+31][0..K) into LDS asynchronously (16B chunks) ---
    int chunksPerCol = K >> 3;                // K/8 chunks of 8 bf16
    int nChunks = 32 * chunksPerCol;
    for (int idx = tid; idx < nChunks; idx += 256) {
        int cc = idx / chunksPerCol;
        int kq = idx - cc * chunksPerCol;
        const __bf16* gp = W + (size_t)(col0 + cc) * ldw + kq * 8;
        unsigned ldsoff = (unsigned)(uintptr_t)(&shW[cc * K + kq * 8]);
        unsigned long long ga = (unsigned long long)(uintptr_t)gp;
        asm volatile("global_load_async_to_lds_b128 %0, %1, off"
                     :: "v"(ldsoff), "v"(ga) : "memory");
    }
    asm volatile("s_wait_asynccnt 0x0" ::: "memory");
    __syncthreads();

    const __bf16* Ap = A + (size_t)(row0 + mrow) * lda;
    v8f acc0 = {};
    v8f acc1 = {};
    for (int k0 = 0; k0 < K; k0 += 32) {
        __builtin_prefetch(Ap + k0 + 32, 0, 0);
        v16bf a, b0, b1;
#pragma unroll
        for (int j = 0; j < 8; ++j) {
            int ka = k0 + 16 * kh + 2 * j;
            a[2 * j]     = Ap[ka];
            a[2 * j + 1] = Ap[ka + 1];
        }
        const __bf16* Bp0 = &shW[mrow * K + k0 + 16 * kh];
        const __bf16* Bp1 = &shW[(mrow + 16) * K + k0 + 16 * kh];
#pragma unroll
        for (int j = 0; j < 16; ++j) { b0[j] = Bp0[j]; b1[j] = Bp1[j]; }
        acc0 = __builtin_amdgcn_wmma_f32_16x16x32_bf16(false, a, false, b0,
                                                       (short)0, acc0, false, false);
        acc1 = __builtin_amdgcn_wmma_f32_16x16x32_bf16(false, a, false, b1,
                                                       (short)0, acc1, false, false);
    }
    int n0 = col0 + mrow;
    int n1 = n0 + 16;
    float bv0 = (bias && n0 < Nvalid) ? bias[n0] : 0.f;
    float bv1 = (bias && n1 < Nvalid) ? bias[n1] : 0.f;
#pragma unroll
    for (int r = 0; r < 8; ++r) {
        int m = row0 + r + 8 * kh;
        if (n0 < Nvalid) {
            float val = acc0[r] + bv0;
            if (outBf16) ((__bf16*)Cout)[(size_t)m * ldc + n0] = f2bf(val);
            else         ((float*)Cout)[(size_t)m * ldc + n0] = val;
        }
        if (n1 < Nvalid) {
            float val = acc1[r] + bv1;
            if (outBf16) ((__bf16*)Cout)[(size_t)m * ldc + n1] = f2bf(val);
            else         ((float*)Cout)[(size_t)m * ldc + n1] = val;
        }
    }
}

// ---------------- per-row featurization ----------------
struct RowArgs {
    const float *x, *ref0, *ref1, *ref2;
    const float *iW1, *ib1, *ig1, *ibn1;
    const float *iW2, *ib2, *ig2, *ibn2;
    const float *iW3, *ib3, *irw;
    const float *scw, *scb;
    const float *g_vec, *b_vec, *g_adj, *b_adj, *g_sym, *b_sym;
    const float *g_m64, *b_m64, *g_s77a, *b_s77a, *g_s77b, *b_s77b;
};

__global__ void k_rowfeat(RowArgs A, const float* __restrict__ cst,
                          float* __restrict__ lowerF, __bf16* __restrict__ lowerB,
                          float* __restrict__ mix) {
    int row = blockIdx.x * blockDim.x + threadIdx.x;
    if (row >= BROWS) return;
    const float* PHI  = cst + OFF_PHI;
    const float* B14c = cst + OFF_B14;
    const float* S27c = cst + OFF_S27;
    const float* P64c = cst + OFF_P64;
    const float* P77Ac = cst + OFF_P77A;
    const float* P77Bc = cst + OFF_P77B;

    float x[7], r0[7], r1[7], r2[7];
    for (int i = 0; i < 7; ++i) x[i] = A.x[row * 7 + i];
    {
        float n0 = 0, n1 = 0, n2 = 0;
        for (int i = 0; i < 7; ++i) { n0 += A.ref0[i]*A.ref0[i]; n1 += A.ref1[i]*A.ref1[i]; n2 += A.ref2[i]*A.ref2[i]; }
        float s0 = rsqrtf(n0), s1 = rsqrtf(n1), s2 = rsqrtf(n2);
        for (int i = 0; i < 7; ++i) { r0[i] = A.ref0[i]*s0; r1[i] = A.ref1[i]*s1; r2[i] = A.ref2[i]*s2; }
    }
    float nx = 0, in0 = 0, in1 = 0, in2 = 0;
    for (int i = 0; i < 7; ++i) { nx += x[i]*x[i]; in0 += x[i]*r0[i]; in1 += x[i]*r1[i]; in2 += x[i]*r2[i]; }
    nx = sqrtf(nx);
    float cr[7];
    for (int k = 0; k < 7; ++k) {
        float s = 0.f;
        for (int i = 0; i < 7; ++i)
            for (int j = 0; j < 7; ++j)
                s += PHI[i*49 + j*7 + k] * x[i] * r0[j];
        cr[k] = s;
    }
    float ncr = 0, trip = 0;
    for (int k = 0; k < 7; ++k) { ncr += cr[k]*cr[k]; trip += cr[k]*r1[k]; }
    ncr = sqrtf(ncr);
    float inv[6] = { nx, in0, in1, in2, ncr, trip };

    {
        float h1[64];
#pragma clang loop unroll(disable)
        for (int o = 0; o < 64; ++o) {
            float t = A.ib1[o];
            for (int i = 0; i < 6; ++i) t += A.iW1[o*6 + i] * inv[i];
            h1[o] = t;
        }
        ln_inplace(h1, 64, A.ig1, A.ibn1);
        for (int o = 0; o < 64; ++o) h1[o] = geluf(h1[o]);
        float h2[32];
#pragma clang loop unroll(disable)
        for (int o = 0; o < 32; ++o) {
            float t = A.ib2[o];
            for (int i = 0; i < 64; ++i) t += A.iW2[o*64 + i] * h1[i];
            h2[o] = t;
        }
        ln_inplace(h2, 32, A.ig2, A.ibn2);
        for (int o = 0; o < 32; ++o) h2[o] = geluf(h2[o]);
        float lg[8], mx = -1e30f;
        for (int o = 0; o < 8; ++o) {
            float t = A.ib3[o] + A.irw[o];
            for (int i = 0; i < 32; ++i) t += A.iW3[o*32 + i] * h2[i];
            lg[o] = t;
            mx = fmaxf(mx, t);
        }
        float den = 0.f;
        for (int o = 0; o < 8; ++o) { lg[o] = expf(lg[o] - mx); den += lg[o]; }
        float idn = 1.f / den;
        for (int o = 0; o < 8; ++o) mix[(size_t)row * 8 + o] = lg[o] * idn;
    }

    float a14[14], b14v[14];
#pragma clang loop unroll(disable)
    for (int m = 0; m < 14; ++m) {
        const float* Bm = B14c + m * 49;
        float ta = 0, tb = 0;
        for (int i = 0; i < 7; ++i)
            for (int j = 0; j < 7; ++j) {
                float w = Bm[i*7 + j];
                ta += w * x[i] * r0[j];
                tb += w * x[i] * r1[j];
            }
        a14[m] = ta; b14v[m] = tb;
    }
    float symA[27];
#pragma clang loop unroll(disable)
    for (int m = 0; m < 27; ++m) {
        const float* Sm = S27c + m * 49;
        float t = 0.f;
        for (int i = 0; i < 7; ++i)
            for (int j = 0; j < 7; ++j) {
                float Sij = 0.5f * (x[i]*r0[j] + x[j]*r0[i]) - ((i == j) ? in0 / 7.f : 0.f);
                t += Sm[i*7 + j] * Sij;
            }
        symA[m] = t;
    }
    float A7[49];
    for (int i = 0; i < 7; ++i)
        for (int j = 0; j < 7; ++j) {
            float t = 0.f;
            for (int m = 0; m < 14; ++m) t += a14[m] * B14c[m*49 + i*7 + j];
            A7[i*7 + j] = t;
        }
    float w7[7];
    for (int i = 0; i < 7; ++i) {
        float t = 0.f;
        for (int j = 0; j < 7; ++j) t += A7[i*7 + j] * x[j];
        w7[i] = t;
    }
    float swx = 0.f;
    for (int i = 0; i < 7; ++i) swx += w7[i] * x[i];
    float sym714[27];
#pragma clang loop unroll(disable)
    for (int m = 0; m < 27; ++m) {
        const float* Sm = S27c + m * 49;
        float t = 0.f;
        for (int i = 0; i < 7; ++i)
            for (int j = 0; j < 7; ++j) {
                float Sij = 0.5f * (w7[i]*x[j] + w7[j]*x[i]) - ((i == j) ? swx / 7.f : 0.f);
                t += Sm[i*7 + j] * Sij;
            }
        sym714[m] = t;
    }
    float m64[64];
#pragma clang loop unroll(disable)
    for (int o = 0; o < 64; ++o) {
        const float* P = P64c + o * 98;
        float t = 0.f;
        for (int i = 0; i < 7; ++i)
            for (int j = 0; j < 14; ++j)
                t += x[i] * a14[j] * P[i*14 + j];
        m64[o] = t;
    }
    float Bm7[49];
    for (int i = 0; i < 7; ++i)
        for (int j = 0; j < 7; ++j) {
            float t = 0.f;
            for (int m = 0; m < 14; ++m) t += b14v[m] * B14c[m*49 + i*7 + j];
            Bm7[i*7 + j] = t;
        }
    float tr = 0.f;
    for (int i = 0; i < 7; ++i)
        for (int k = 0; k < 7; ++k)
            tr += A7[i*7 + k] * Bm7[k*7 + i];
    float adj1414[14], sym1414[27];
    for (int m = 0; m < 14; ++m) adj1414[m] = 0.f;
    for (int m = 0; m < 27; ++m) sym1414[m] = 0.f;
#pragma clang loop unroll(disable)
    for (int i = 0; i < 7; ++i)
        for (int j = 0; j < 7; ++j) {
            float ab = 0.f, ba = 0.f;
            for (int k = 0; k < 7; ++k) {
                ab += A7[i*7 + k] * Bm7[k*7 + j];
                ba += Bm7[i*7 + k] * A7[k*7 + j];
            }
            float d = ab - ba;
            float s = 0.5f * (ab + ba) - ((i == j) ? tr / 7.f : 0.f);
            for (int m = 0; m < 14; ++m) adj1414[m] += B14c[m*49 + i*7 + j] * d;
            for (int m = 0; m < 27; ++m) sym1414[m] += S27c[m*49 + i*7 + j] * s;
        }
    float scal1414 = 0.f;
    for (int m = 0; m < 14; ++m) scal1414 += a14[m] * b14v[m];
    float s77a[77], s77b[77];
#pragma clang loop unroll(disable)
    for (int o = 0; o < 77; ++o) {
        const float* Pa = P77Ac + o * 196;
        const float* Pb = P77Bc + o * 196;
        float ta = 0.f, tb = 0.f;
        for (int i = 0; i < 14; ++i) {
            float ai = a14[i];
            for (int j = 0; j < 14; ++j) {
                float t = ai * b14v[j];
                ta += t * Pa[i*14 + j];
                tb += t * Pb[i*14 + j];
            }
        }
        s77a[o] = ta; s77b[o] = tb;
    }
    float irvec[7];
    for (int i = 0; i < 7; ++i) irvec[i] = x[i];
    ln_inplace(irvec, 7, A.g_vec, A.b_vec);
    for (int i = 0; i < 7; ++i) irvec[i] += 0.1f * w7[i];

    float scw = A.scw[0], scb = A.scb[0];
    float irsc = 0.5f * ((in0 * scw + scb) + (in2 * scw + scb)) + 0.1f * scal1414;

    float iradj[14], tmp14[14];
    for (int i = 0; i < 14; ++i) { iradj[i] = a14[i]; tmp14[i] = adj1414[i]; }
    ln_inplace(iradj, 14, A.g_adj, A.b_adj);
    ln_inplace(tmp14, 14, A.g_adj, A.b_adj);
    for (int i = 0; i < 14; ++i) iradj[i] += 0.1f * tmp14[i];

    float irsym[27], t27a[27], t27b[27];
    for (int i = 0; i < 27; ++i) { irsym[i] = symA[i]; t27a[i] = sym1414[i]; t27b[i] = sym714[i]; }
    ln_inplace(irsym, 27, A.g_sym, A.b_sym);
    ln_inplace(t27a, 27, A.g_sym, A.b_sym);
    ln_inplace(t27b, 27, A.g_sym, A.b_sym);
    for (int i = 0; i < 27; ++i) irsym[i] += 0.1f * (t27a[i] + t27b[i]);

    ln_inplace(m64, 64, A.g_m64, A.b_m64);
    ln_inplace(s77a, 77, A.g_s77a, A.b_s77a);
    ln_inplace(s77b, 77, A.g_s77b, A.b_s77b);

    float* L = lowerF + (size_t)row * 272;
    __bf16* LB = lowerB + (size_t)row * 288;
    int c = 0;
    for (int i = 0; i < 7; ++i)  { L[c] = irvec[i]; LB[c] = f2bf(irvec[i]); ++c; }
    L[c] = irsc; LB[c] = f2bf(irsc); ++c;
    for (int i = 0; i < 14; ++i) { L[c] = iradj[i]; LB[c] = f2bf(iradj[i]); ++c; }
    for (int i = 0; i < 27; ++i) { L[c] = irsym[i]; LB[c] = f2bf(irsym[i]); ++c; }
    for (int i = 0; i < 64; ++i) { L[c] = m64[i];  LB[c] = f2bf(m64[i]);  ++c; }
    for (int i = 0; i < 77; ++i) { L[c] = s77a[i]; LB[c] = f2bf(s77a[i]); ++c; }
    for (int i = 0; i < 77; ++i) { L[c] = s77b[i]; LB[c] = f2bf(s77b[i]); ++c; }
    for (int t = 267; t < 288; ++t) LB[t] = f2bf(0.f);
}

// ---------------- LN(h189) + build padded tokens ----------------
__global__ void k_build(const float* __restrict__ lowerF, float* __restrict__ h189,
                        const float* __restrict__ gH, const float* __restrict__ bH,
                        __bf16* __restrict__ padded) {
    int row = blockIdx.x * blockDim.x + threadIdx.x;
    if (row >= BROWS) return;
    float* H = h189 + (size_t)row * 192;
    float m = 0.f;
    for (int c = 0; c < 189; ++c) m += H[c];
    m /= 189.f;
    float var = 0.f;
    for (int c = 0; c < 189; ++c) { float d = H[c] - m; var += d * d; }
    var /= 189.f;
    float is = rsqrtf(var + 1e-5f);
    __bf16* P = padded + (size_t)row * 512;
    const float* L = lowerF + (size_t)row * 272;
    for (int c = 0; c < 267; ++c) P[c] = f2bf(L[c]);
    for (int c = 0; c < 189; ++c) {
        float v = (H[c] - m) * is * gH[c] + bH[c];
        H[c] = v;
        P[267 + c] = f2bf(v);
    }
    for (int c = 456; c < 512; ++c) P[c] = f2bf(0.f);
}

// ---------------- attention core (softmax over 8 tokens, 4 heads of 16) ----------------
__global__ void k_attn(const __bf16* __restrict__ q, const __bf16* __restrict__ k,
                       const __bf16* __restrict__ v, __bf16* __restrict__ ao) {
    int t = blockIdx.x * blockDim.x + threadIdx.x;   // B*32
    if (t >= BROWS * 32) return;
    int b = t >> 5;
    int h = (t >> 3) & 3;
    int qt = t & 7;
    const __bf16* qp = q + ((size_t)(b * 8 + qt)) * 64 + h * 16;
    float qv[16];
    for (int d = 0; d < 16; ++d) qv[d] = bf2f(qp[d]);
    float sc[8], mx = -1e30f;
    for (int kt = 0; kt < 8; ++kt) {
        const __bf16* kp = k + ((size_t)(b * 8 + kt)) * 64 + h * 16;
        float s = 0.f;
        for (int d = 0; d < 16; ++d) s += qv[d] * bf2f(kp[d]);
        s *= 0.25f;
        sc[kt] = s;
        mx = fmaxf(mx, s);
    }
    float den = 0.f;
    for (int kt = 0; kt < 8; ++kt) { sc[kt] = expf(sc[kt] - mx); den += sc[kt]; }
    float idn = 1.f / den;
    float o[16];
    for (int d = 0; d < 16; ++d) o[d] = 0.f;
    for (int kt = 0; kt < 8; ++kt) {
        const __bf16* vp = v + ((size_t)(b * 8 + kt)) * 64 + h * 16;
        float w = sc[kt] * idn;
        for (int d = 0; d < 16; ++d) o[d] += w * bf2f(vp[d]);
    }
    __bf16* op = ao + ((size_t)(b * 8 + qt)) * 64 + h * 16;
    for (int d = 0; d < 16; ++d) op[d] = f2bf(o[d]);
}

// ---------------- residual + mixing scale -> x1 bf16 ----------------
__global__ void k_combine(const float* __restrict__ lowerF, const float* __restrict__ h189,
                          const float* __restrict__ attn, const float* __restrict__ mix,
                          __bf16* __restrict__ x1) {
    size_t t = (size_t)blockIdx.x * blockDim.x + threadIdx.x;
    if (t >= (size_t)BROWS * 480) return;
    int b = (int)(t / 480);
    int c = (int)(t % 480);
    __bf16* out = x1 + (size_t)b * 480 + c;
    if (c >= 456) { *out = f2bf(0.f); return; }
    float base = (c < 267) ? lowerF[(size_t)b * 272 + c] : h189[(size_t)b * 192 + (c - 267)];
    float a = base + 0.1f * attn[(size_t)b * 512 + c];
    int seg = (c < 7) ? 0 : (c < 8) ? 1 : (c < 22) ? 2 : (c < 49) ? 3 :
              (c < 113) ? 4 : (c < 190) ? 5 : (c < 267) ? 6 : 7;
    float mval = mix[(size_t)b * 8 + seg];
    *out = f2bf(a * (1.0f + 0.1f * mval));
}

// ---------------- LN(256) + gelu -> bf16 ----------------
__global__ void k_lngelu(const float* __restrict__ pre, const float* __restrict__ g,
                         const float* __restrict__ bb, __bf16* __restrict__ h) {
    int row = blockIdx.x * blockDim.x + threadIdx.x;
    if (row >= BROWS) return;
    const float* P = pre + (size_t)row * 256;
    float m = 0.f;
    for (int c = 0; c < 256; ++c) m += P[c];
    m /= 256.f;
    float var = 0.f;
    for (int c = 0; c < 256; ++c) { float d = P[c] - m; var += d * d; }
    var /= 256.f;
    float is = rsqrtf(var + 1e-5f);
    for (int c = 0; c < 256; ++c) {
        float v = (P[c] - m) * is * g[c] + bb[c];
        h[(size_t)row * 256 + c] = f2bf(geluf(v));
    }
}

// ---------------- host launcher ----------------
extern "C" void kernel_launch(void* const* d_in, const int* in_sizes, int n_in,
                              void* d_out, int out_size, void* d_ws, size_t ws_size,
                              hipStream_t stream) {
    (void)in_sizes; (void)n_in; (void)out_size; (void)ws_size;
    char* ws = (char*)d_ws;
    float* cst = (float*)(ws + O_CST);

    #define IN(i) ((const float*)d_in[(i)])

    // constants + weight casts
    k_setup<<<1, 1, 0, stream>>>(cst);
    {
        int threads = 256;
        k_castpad<<<(192 * 288 + threads - 1) / threads, threads, 0, stream>>>(IN(31), (__bf16*)(ws + O_WHPB), 189, 267, 192, 288);
        k_castpad<<<(64 * 64 + threads - 1) / threads, threads, 0, stream>>>(IN(32), (__bf16*)(ws + O_WQB), 64, 64, 64, 64);
        k_castpad<<<(64 * 64 + threads - 1) / threads, threads, 0, stream>>>(IN(33), (__bf16*)(ws + O_WKB), 64, 64, 64, 64);
        k_castpad<<<(64 * 64 + threads - 1) / threads, threads, 0, stream>>>(IN(34), (__bf16*)(ws + O_WVB), 64, 64, 64, 64);
        k_castpad<<<(64 * 64 + threads - 1) / threads, threads, 0, stream>>>(IN(35), (__bf16*)(ws + O_WOB), 64, 64, 64, 64);
        k_castpad<<<(256 * 480 + threads - 1) / threads, threads, 0, stream>>>(IN(40), (__bf16*)(ws + O_W1B), 256, 456, 256, 480);
        k_castpad<<<(128 * 256 + threads - 1) / threads, threads, 0, stream>>>(IN(44), (__bf16*)(ws + O_W2B), 128, 256, 128, 256);
    }

    // per-row featurization
    RowArgs ra;
    ra.x = IN(0); ra.ref0 = IN(1); ra.ref1 = IN(2); ra.ref2 = IN(3);
    ra.iW1 = IN(4); ra.ib1 = IN(5); ra.ig1 = IN(6); ra.ibn1 = IN(7);
    ra.iW2 = IN(8); ra.ib2 = IN(9); ra.ig2 = IN(10); ra.ibn2 = IN(11);
    ra.iW3 = IN(12); ra.ib3 = IN(13); ra.irw = IN(14);
    ra.scw = IN(15); ra.scb = IN(16);
    ra.g_vec = IN(17); ra.b_vec = IN(18);
    ra.g_adj = IN(19); ra.b_adj = IN(20);
    ra.g_sym = IN(21); ra.b_sym = IN(22);
    ra.g_m64 = IN(23); ra.b_m64 = IN(24);
    ra.g_s77a = IN(25); ra.b_s77a = IN(26);
    ra.g_s77b = IN(27); ra.b_s77b = IN(28);
    k_rowfeat<<<BROWS / 128, 128, 0, stream>>>(ra, cst,
        (float*)(ws + O_LOWF), (__bf16*)(ws + O_LOWB), (float*)(ws + O_MIX));

    // h189 = lower @ Whp^T    (M=B, K=288, N=192/189)
    k_gemm_bf16<<<dim3(6, BROWS / 128), 256, 0, stream>>>(
        (const __bf16*)(ws + O_LOWB), 288, (const __bf16*)(ws + O_WHPB), 288,
        nullptr, ws + O_H189, 192, 288, 189, 0);

    // LN + token assembly
    k_build<<<BROWS / 128, 128, 0, stream>>>(
        (float*)(ws + O_LOWF), (float*)(ws + O_H189), IN(29), IN(30),
        (__bf16*)(ws + O_PAD));

    // Q/K/V projections  (M=8B, K=64, N=64)
    k_gemm_bf16<<<dim3(2, BROWS * 8 / 128), 256, 0, stream>>>(
        (const __bf16*)(ws + O_PAD), 64, (const __bf16*)(ws + O_WQB), 64,
        IN(36), ws + O_Q, 64, 64, 64, 1);
    k_gemm_bf16<<<dim3(2, BROWS * 8 / 128), 256, 0, stream>>>(
        (const __bf16*)(ws + O_PAD), 64, (const __bf16*)(ws + O_WKB), 64,
        IN(37), ws + O_KB, 64, 64, 64, 1);
    k_gemm_bf16<<<dim3(2, BROWS * 8 / 128), 256, 0, stream>>>(
        (const __bf16*)(ws + O_PAD), 64, (const __bf16*)(ws + O_WVB), 64,
        IN(38), ws + O_V, 64, 64, 64, 1);

    // attention softmax core
    k_attn<<<(BROWS * 32) / 256, 256, 0, stream>>>(
        (const __bf16*)(ws + O_Q), (const __bf16*)(ws + O_KB),
        (const __bf16*)(ws + O_V), (__bf16*)(ws + O_AO));

    // output projection
    k_gemm_bf16<<<dim3(2, BROWS * 8 / 128), 256, 0, stream>>>(
        (const __bf16*)(ws + O_AO), 64, (const __bf16*)(ws + O_WOB), 64,
        IN(39), ws + O_ATT, 64, 64, 64, 0);

    // residual + mixing scale
    k_combine<<<(unsigned)(((size_t)BROWS * 480 + 255) / 256), 256, 0, stream>>>(
        (const float*)(ws + O_LOWF), (const float*)(ws + O_H189),
        (const float*)(ws + O_ATT), (const float*)(ws + O_MIX),
        (__bf16*)(ws + O_X1));

    // out_W1 (M=B, K=480, N=256)
    k_gemm_bf16<<<dim3(8, BROWS / 128), 256, 0, stream>>>(
        (const __bf16*)(ws + O_X1), 480, (const __bf16*)(ws + O_W1B), 480,
        IN(41), ws + O_PRE1, 256, 480, 256, 0);

    // LN + gelu
    k_lngelu<<<BROWS / 128, 128, 0, stream>>>(
        (const float*)(ws + O_PRE1), IN(42), IN(43), (__bf16*)(ws + O_H256));

    // out_W2 (M=B, K=256, N=128) -> d_out
    k_gemm_bf16<<<dim3(4, BROWS / 128), 256, 0, stream>>>(
        (const __bf16*)(ws + O_H256), 256, (const __bf16*)(ws + O_W2B), 256,
        IN(45), d_out, 128, 256, 128, 0);

    #undef IN
}
